// TransformerEncoder_22497038696852
// MI455X (gfx1250) — matrix operand order repaired and checked by reading, not compiled, not run on hardware
//
#include <hip/hip_runtime.h>
#include <hip/hip_bf16.h>
#include <math.h>

// ---------------------------------------------------------------------------
// T5-style shared-weight transformer encoder for MI455X (gfx1250, wave32).
// All GEMMs + attention matmuls run on V_WMMA_F32_16X16X32_BF16.
// ---------------------------------------------------------------------------

typedef __bf16 bf16_t;
typedef __attribute__((ext_vector_type(16))) __bf16 v16bf;
typedef __attribute__((ext_vector_type(8)))  float  v8f;

#define LAYERS 12
#define HEADS  12
#define HIDDEN 768
#define FFDIM  3072
#define SEQ    512
#define BATCH  8
#define DHEAD  64
#define MTOT   (BATCH * SEQ) /* 4096 */

union FragLd { v16bf v; float4 q[2]; };

// 16-bit A-matrix 16x32 layout: lanes 0-15 -> row M=lane, elems 0..7 = K 0..7,
// elems 8..15 = K 16..23; lanes 16-31 -> same rows, K shifted by 8.
__device__ __forceinline__ v16bf load_frag_a(const bf16_t* base, int ld, int lane) {
  const int r  = lane & 15;
  const int kh = (lane >> 4) * 8;
  FragLd f;
  const bf16_t* p = base + (size_t)r * ld + kh;
  f.q[0] = *reinterpret_cast<const float4*>(p);
  f.q[1] = *reinterpret_cast<const float4*>(p + 16);
  return f.v;
}

// 16-bit B-matrix 32x16 layout (from transposed [N,K] storage): lane -> col N,
// elems 0..7 = K 0..7, elems 8..15 = K 16..23; lanes 16-31 K shifted by 8.
__device__ __forceinline__ v16bf load_frag_b(const bf16_t* base, int ld, int lane) {
  const int n  = lane & 15;
  const int kh = (lane >> 4) * 8;
  FragLd f;
  const bf16_t* p = base + (size_t)n * ld + kh;
  f.q[0] = *reinterpret_cast<const float4*>(p);
  f.q[1] = *reinterpret_cast<const float4*>(p + 16);
  return f.v;
}

__device__ __forceinline__ v8f wmma_bf16(v16bf a, v16bf b, v8f c) {
  return __builtin_amdgcn_wmma_f32_16x16x32_bf16(
      /*neg_a=*/false, a, /*neg_b=*/false, b,
      /*c_mod=*/(short)0, c, /*reuse_a=*/false, /*reuse_b=*/false);
}

__device__ __forceinline__ float gelu_exact(float x) {
  return 0.5f * x * (1.0f + erff(x * 0.70710678118654752f));
}

// ---------------------------------------------------------------------------
// Generic bf16 WMMA GEMM: C[M,N] = act((A[M,K] @ Bt[N,K]^T + bias) * out_scale)
// out_mode: 0 = fp32 row-major, 1 = bf16 row-major,
//           2 = bf16 V-transposed: out[b][n][s] with m = b*SEQ + s, N == HIDDEN
// Block = 8 waves tiling 128x128; each wave owns a 32x64 tile (2x4 accums),
// so each K=32 step does 8 WMMAs per 6 fragment loads (1.33x better fragment
// reuse than a 32x32 tile -> keeps the WMMA pipe fed from the L0 path).
// ---------------------------------------------------------------------------
__global__ __launch_bounds__(256) void gemm_bf16_wmma(
    const bf16_t* __restrict__ A, const bf16_t* __restrict__ Bt,
    const float* __restrict__ bias, void* __restrict__ out,
    int M, int N, int K, int out_mode, int act, float out_scale)
{
  const int lane = threadIdx.x & 31;
  const int w    = threadIdx.x >> 5;
  const int m0   = blockIdx.y * 128 + (w >> 1) * 32;
  const int n0   = blockIdx.x * 128 + (w & 1) * 64;

  v8f acc[2][4] = {};
  const bf16_t* a0p = A + (size_t)m0 * K;
  const bf16_t* a1p = a0p + (size_t)16 * K;
  const bf16_t* bp0 = Bt + (size_t)n0 * K;
  const bf16_t* bp1 = bp0 + (size_t)16 * K;
  const bf16_t* bp2 = bp0 + (size_t)32 * K;
  const bf16_t* bp3 = bp0 + (size_t)48 * K;

  for (int k0 = 0; k0 < K; k0 += 32) {
    v16bf a0 = load_frag_a(a0p + k0, K, lane);
    v16bf a1 = load_frag_a(a1p + k0, K, lane);
    v16bf b0 = load_frag_b(bp0 + k0, K, lane);
    v16bf b1 = load_frag_b(bp1 + k0, K, lane);
    v16bf b2 = load_frag_b(bp2 + k0, K, lane);
    v16bf b3 = load_frag_b(bp3 + k0, K, lane);
    acc[0][0] = wmma_bf16(a0, b0, acc[0][0]);
    acc[0][1] = wmma_bf16(a0, b1, acc[0][1]);
    acc[0][2] = wmma_bf16(a0, b2, acc[0][2]);
    acc[0][3] = wmma_bf16(a0, b3, acc[0][3]);
    acc[1][0] = wmma_bf16(a1, b0, acc[1][0]);
    acc[1][1] = wmma_bf16(a1, b1, acc[1][1]);
    acc[1][2] = wmma_bf16(a1, b2, acc[1][2]);
    acc[1][3] = wmma_bf16(a1, b3, acc[1][3]);
  }

  float*  outf = (float*)out;
  bf16_t* outb = (bf16_t*)out;
  const int nlo = lane & 15;
  const int mhi = (lane >> 4) * 8;
#pragma unroll
  for (int i = 0; i < 2; ++i) {
#pragma unroll
    for (int j = 0; j < 4; ++j) {
      const int mmB = m0 + i * 16 + mhi;
      const int nn  = n0 + j * 16 + nlo;
      const float bv = bias[nn];
#pragma unroll
      for (int r = 0; r < 8; ++r) {
        float vv = (acc[i][j][r] + bv) * out_scale;
        if (act == 1) vv = gelu_exact(vv);
        const int mm = mmB + r;
        if (out_mode == 0) {
          outf[(size_t)mm * N + nn] = vv;
        } else if (out_mode == 1) {
          outb[(size_t)mm * N + nn] = (bf16_t)vv;
        } else {  // V-transposed store for attention B-fragments
          const int bb = mm >> 9;       // / SEQ
          const int ss = mm & (SEQ - 1);
          outb[((size_t)bb * HIDDEN + nn) * SEQ + ss] = (bf16_t)vv;
        }
      }
    }
  }
}

// ---------------------------------------------------------------------------
// Attention: one block per (batch, head, 16-row q tile).
// Phase 1: scores = scaled-Q @ K^T + rel_bias + mask  (WMMA, 8 waves)
// Phase 2: softmax over S=512 in LDS (16 lanes per row, shuffle reductions)
// Phase 3: ctx = P @ V  (WMMA, A-fragments from LDS, B from transposed V)
// ---------------------------------------------------------------------------
__global__ __launch_bounds__(256) void attention_wmma(
    const bf16_t* __restrict__ Q, const bf16_t* __restrict__ Kmat,
    const bf16_t* __restrict__ Vt, const float* __restrict__ rel_emb,
    const int* __restrict__ bucket, const float* __restrict__ mask,
    bf16_t* __restrict__ Ctx)
{
  __shared__ float  sc[16 * SEQ];   // 32 KB fp32 scores
  __shared__ bf16_t pr[16 * SEQ];   // 16 KB bf16 probs

  const int lane = threadIdx.x & 31;
  const int w    = threadIdx.x >> 5;
  const int bid  = blockIdx.x;
  const int qt   = bid & 31;                // SEQ/16 = 32 q tiles
  const int h    = (bid >> 5) % HEADS;
  const int b    = bid / (32 * HEADS);
  const int q0   = qt * 16;

  const bf16_t* qbase = Q + ((size_t)(b * SEQ + q0) * HIDDEN + h * DHEAD);

  // ---- Phase 1: Q K^T tiles -> LDS scores ----
  for (int kt = w; kt < 32; kt += 8) {
    const bf16_t* kbase = Kmat + ((size_t)(b * SEQ + kt * 16) * HIDDEN + h * DHEAD);
    v8f acc = {};
#pragma unroll
    for (int c = 0; c < DHEAD; c += 32) {
      v16bf af  = load_frag_a(qbase + c, HIDDEN, lane);
      v16bf bfr = load_frag_b(kbase + c, HIDDEN, lane);
      acc = wmma_bf16(af, bfr, acc);
    }
    const int nn = kt * 16 + (lane & 15);
    const int mh = (lane >> 4) * 8;
    const float mv = mask[b * SEQ + nn];
#pragma unroll
    for (int r = 0; r < 8; ++r) {
      const int mm = mh + r;
      const int bk = bucket[(q0 + mm) * SEQ + nn];
      sc[mm * SEQ + nn] = acc[r] + rel_emb[bk * HEADS + h] + mv;
    }
  }
  __syncthreads();

  // ---- Phase 2: softmax (16 threads per row) ----
  {
    const int row = threadIdx.x >> 4;
    const int j   = threadIdx.x & 15;
    float mx = -1e30f;
    for (int cidx = j; cidx < SEQ; cidx += 16)
      mx = fmaxf(mx, sc[row * SEQ + cidx]);
#pragma unroll
    for (int off = 8; off; off >>= 1) mx = fmaxf(mx, __shfl_xor(mx, off, 16));
    float sum = 0.f;
    for (int cidx = j; cidx < SEQ; cidx += 16) {
      const float e = __expf(sc[row * SEQ + cidx] - mx);
      sc[row * SEQ + cidx] = e;
      sum += e;
    }
#pragma unroll
    for (int off = 8; off; off >>= 1) sum += __shfl_xor(sum, off, 16);
    const float inv = 1.0f / sum;
    for (int cidx = j; cidx < SEQ; cidx += 16)
      pr[row * SEQ + cidx] = (bf16_t)(sc[row * SEQ + cidx] * inv);
  }
  __syncthreads();

  // ---- Phase 3: ctx = P @ V (waves 0..3, one 16-feature tile each) ----
  if (w < 4) {
    const bf16_t* vbase = Vt + ((size_t)(b * HIDDEN + h * DHEAD + w * 16)) * SEQ;
    v8f acc = {};
    for (int c = 0; c < SEQ; c += 32) {
      v16bf af  = load_frag_a(pr + c, SEQ, lane);     // ds_load path
      v16bf bfr = load_frag_b(vbase + c, SEQ, lane);
      acc = wmma_bf16(af, bfr, acc);
    }
    const int nn = lane & 15;
    const int mh = (lane >> 4) * 8;
#pragma unroll
    for (int r = 0; r < 8; ++r) {
      const int mm = mh + r;
      Ctx[(size_t)(b * SEQ + q0 + mm) * HIDDEN + h * DHEAD + w * 16 + nn] = (bf16_t)acc[r];
    }
  }
}

// ---------------------------------------------------------------------------
// Fused residual + LayerNorm; writes fp32 hidden and bf16 copy for next GEMM.
// One wave per 768-wide row (24 elements/lane).
// ---------------------------------------------------------------------------
__global__ __launch_bounds__(256) void ln_residual(
    const float* __restrict__ x, const float* __restrict__ res,
    const float* __restrict__ g, const float* __restrict__ bta,
    float* __restrict__ out, bf16_t* __restrict__ outb)
{
  const int row  = blockIdx.x * 8 + (threadIdx.x >> 5);
  const int lane = threadIdx.x & 31;
  const float* px = x   + (size_t)row * HIDDEN;
  const float* pres = res + (size_t)row * HIDDEN;
  float vals[24];
  float s = 0.f;
#pragma unroll
  for (int i = 0; i < 24; ++i) {
    vals[i] = px[lane + i * 32] + pres[lane + i * 32];
    s += vals[i];
  }
#pragma unroll
  for (int off = 16; off; off >>= 1) s += __shfl_xor(s, off, 32);
  const float mean = s * (1.0f / HIDDEN);
  float vs = 0.f;
#pragma unroll
  for (int i = 0; i < 24; ++i) { const float d = vals[i] - mean; vs += d * d; }
#pragma unroll
  for (int off = 16; off; off >>= 1) vs += __shfl_xor(vs, off, 32);
  const float inv = rsqrtf(vs * (1.0f / HIDDEN) + 1e-6f);
#pragma unroll
  for (int i = 0; i < 24; ++i) {
    const int col = lane + i * 32;
    const float o = g[col] * (vals[i] - mean) * inv + bta[col];
    out[(size_t)row * HIDDEN + col]  = o;
    outb[(size_t)row * HIDDEN + col] = (bf16_t)o;
  }
}

// ---------------------------------------------------------------------------
// Setup kernels
// ---------------------------------------------------------------------------
__global__ void cast_transpose(const float* __restrict__ W, bf16_t* __restrict__ Wt,
                               int K, int N)
{
  const int idx = blockIdx.x * blockDim.x + threadIdx.x;
  if (idx >= K * N) return;
  const int k = idx / N, n = idx % N;
  Wt[(size_t)n * K + k] = (bf16_t)W[idx];
}

__global__ void cast_bf16(const float* __restrict__ x, bf16_t* __restrict__ y, int n) {
  const int idx = blockIdx.x * blockDim.x + threadIdx.x;
  if (idx < n) y[idx] = (bf16_t)x[idx];
}

__global__ void build_bucket(int* __restrict__ bucket) {
  const int idx = blockIdx.x * blockDim.x + threadIdx.x;
  if (idx >= SEQ * SEQ) return;
  const int qi = idx / SEQ, ki = idx % SEQ;
  const int rel = ki - qi;
  int bkt = (rel > 0) ? 16 : 0;          // num_buckets/2
  const int a = rel < 0 ? -rel : rel;
  int val;
  if (a < 8) {                           // max_exact = 8
    val = a;
  } else {
    const float lf = __logf((float)a * 0.125f) * (1.0f / __logf(16.0f)) * 8.0f;
    const int vi = 8 + (int)lf;
    val = vi < 15 ? vi : 15;
  }
  bucket[idx] = bkt + val;
}

__global__ void build_mask(const int* __restrict__ seg, float* __restrict__ mask) {
  const int idx = blockIdx.x * blockDim.x + threadIdx.x;
  if (idx >= BATCH * SEQ) return;
  mask[idx] = (seg[idx] > 0) ? 0.0f : -10000.0f;
}

// ---------------------------------------------------------------------------
// Host driver
// ---------------------------------------------------------------------------
extern "C" void kernel_launch(void* const* d_in, const int* in_sizes, int n_in,
                              void* d_out, int out_size, void* d_ws, size_t ws_size,
                              hipStream_t stream)
{
  const float* emb     = (const float*)d_in[0];
  const int*   seg     = (const int*)  d_in[1];
  const float* rel_emb = (const float*)d_in[2];
  const float* Wq = (const float*)d_in[3];  const float* bq = (const float*)d_in[4];
  const float* Wk = (const float*)d_in[5];  const float* bk = (const float*)d_in[6];
  const float* Wv = (const float*)d_in[7];  const float* bv = (const float*)d_in[8];
  const float* Wo = (const float*)d_in[9];  const float* bo = (const float*)d_in[10];
  const float* ln1_g = (const float*)d_in[11]; const float* ln1_b = (const float*)d_in[12];
  const float* W1 = (const float*)d_in[13]; const float* b1 = (const float*)d_in[14];
  const float* W2 = (const float*)d_in[15]; const float* b2 = (const float*)d_in[16];
  const float* ln2_g = (const float*)d_in[17]; const float* ln2_b = (const float*)d_in[18];

  char* ws = (char*)d_ws;
  size_t off = 0;
  auto alloc = [&](size_t bytes) -> void* {
    void* p = ws + off;
    off += bytes;
    off = (off + 255) & ~(size_t)255;
    return p;
  };

  bf16_t* wq_t = (bf16_t*)alloc((size_t)HIDDEN * HIDDEN * 2);
  bf16_t* wk_t = (bf16_t*)alloc((size_t)HIDDEN * HIDDEN * 2);
  bf16_t* wv_t = (bf16_t*)alloc((size_t)HIDDEN * HIDDEN * 2);
  bf16_t* wo_t = (bf16_t*)alloc((size_t)HIDDEN * HIDDEN * 2);
  bf16_t* w1_t = (bf16_t*)alloc((size_t)HIDDEN * FFDIM * 2);
  bf16_t* w2_t = (bf16_t*)alloc((size_t)HIDDEN * FFDIM * 2);
  float*  hidden    = (float*) alloc((size_t)MTOT * HIDDEN * 4);
  bf16_t* hidden_bf = (bf16_t*)alloc((size_t)MTOT * HIDDEN * 2);
  bf16_t* qb   = (bf16_t*)alloc((size_t)MTOT * HIDDEN * 2);
  bf16_t* kb   = (bf16_t*)alloc((size_t)MTOT * HIDDEN * 2);
  bf16_t* vt   = (bf16_t*)alloc((size_t)MTOT * HIDDEN * 2);
  bf16_t* ctxb = (bf16_t*)alloc((size_t)MTOT * HIDDEN * 2);
  float*  tmp  = (float*) alloc((size_t)MTOT * HIDDEN * 4);
  bf16_t* ffb  = (bf16_t*)alloc((size_t)MTOT * FFDIM * 2);
  int*    bucket = (int*)  alloc((size_t)SEQ * SEQ * 4);
  float*  mask   = (float*)alloc((size_t)BATCH * SEQ * 4);
  (void)ws_size; (void)in_sizes; (void)n_in;

  // --- one-time (per call) setup ---
  const int whh = HIDDEN * HIDDEN, whf = HIDDEN * FFDIM;
  cast_transpose<<<(whh + 255) / 256, 256, 0, stream>>>(Wq, wq_t, HIDDEN, HIDDEN);
  cast_transpose<<<(whh + 255) / 256, 256, 0, stream>>>(Wk, wk_t, HIDDEN, HIDDEN);
  cast_transpose<<<(whh + 255) / 256, 256, 0, stream>>>(Wv, wv_t, HIDDEN, HIDDEN);
  cast_transpose<<<(whh + 255) / 256, 256, 0, stream>>>(Wo, wo_t, HIDDEN, HIDDEN);
  cast_transpose<<<(whf + 255) / 256, 256, 0, stream>>>(W1, w1_t, HIDDEN, FFDIM);
  cast_transpose<<<(whf + 255) / 256, 256, 0, stream>>>(W2, w2_t, FFDIM, HIDDEN);
  build_bucket<<<(SEQ * SEQ + 255) / 256, 256, 0, stream>>>(bucket);
  build_mask<<<(BATCH * SEQ + 255) / 256, 256, 0, stream>>>(seg, mask);
  hipMemcpyAsync(hidden, emb, (size_t)MTOT * HIDDEN * 4, hipMemcpyDeviceToDevice, stream);
  cast_bf16<<<(MTOT * HIDDEN + 255) / 256, 256, 0, stream>>>(hidden, hidden_bf, MTOT * HIDDEN);

  const dim3 gemmH(HIDDEN / 128, MTOT / 128);  // (6, 32)
  const dim3 gemmF(FFDIM  / 128, MTOT / 128);  // (24, 32)
  const float qscale = 0.125f;                 // 1/sqrt(64)

  for (int layer = 0; layer < LAYERS; ++layer) {
    gemm_bf16_wmma<<<gemmH, 256, 0, stream>>>(hidden_bf, wq_t, bq, qb,
                                              MTOT, HIDDEN, HIDDEN, 1, 0, qscale);
    gemm_bf16_wmma<<<gemmH, 256, 0, stream>>>(hidden_bf, wk_t, bk, kb,
                                              MTOT, HIDDEN, HIDDEN, 1, 0, 1.0f);
    gemm_bf16_wmma<<<gemmH, 256, 0, stream>>>(hidden_bf, wv_t, bv, vt,
                                              MTOT, HIDDEN, HIDDEN, 2, 0, 1.0f);
    attention_wmma<<<BATCH * HEADS * (SEQ / 16), 256, 0, stream>>>(
        qb, kb, vt, rel_emb, bucket, mask, ctxb);
    gemm_bf16_wmma<<<gemmH, 256, 0, stream>>>(ctxb, wo_t, bo, tmp,
                                              MTOT, HIDDEN, HIDDEN, 0, 0, 1.0f);
    ln_residual<<<MTOT / 8, 256, 0, stream>>>(tmp, hidden, ln1_g, ln1_b,
                                              hidden, hidden_bf);
    gemm_bf16_wmma<<<gemmF, 256, 0, stream>>>(hidden_bf, w1_t, b1, ffb,
                                              MTOT, FFDIM, HIDDEN, 1, 1, 1.0f);
    gemm_bf16_wmma<<<gemmH, 256, 0, stream>>>(ffb, w2_t, b2, tmp,
                                              MTOT, HIDDEN, FFDIM, 0, 0, 1.0f);
    ln_residual<<<MTOT / 8, 256, 0, stream>>>(tmp, hidden, ln2_g, ln2_b,
                                              hidden, hidden_bf);
  }

  hipMemcpyAsync(d_out, hidden, (size_t)out_size * sizeof(float),
                 hipMemcpyDeviceToDevice, stream);
}
